// MultiHeadSelfAttention_17128329576523
// MI455X (gfx1250) — matrix operand order repaired and checked by reading, not compiled, not run on hardware
//
#include <hip/hip_runtime.h>

// ---------------------------------------------------------------------------
// MHA forward for MI455X (gfx1250, wave32).  All matmuls on v_wmma_f32_16x16x32_bf16.
// Pipeline: 3x proj GEMM (fp32 x -> bf16 Q/K/V head-split) -> flash attention
// (bf16 QK^T + PV, fp32 online softmax, async V staging to LDS, LDS P re-layout)
// -> out-proj GEMM (fp32 out).
// ---------------------------------------------------------------------------

typedef __attribute__((ext_vector_type(16))) __bf16 v16bf;
typedef __attribute__((ext_vector_type(8)))  float  v8f;

#define BATCH  4
#define SEQ    2048
#define DMODEL 1024
#define NHEAD  16
#define HDIM   64

struct FragBF {
  union { v16bf v; unsigned u[8]; };
};

static __device__ __forceinline__ unsigned short f32_to_bf16(float f) {
  unsigned u = __float_as_uint(f);
  u = u + 0x7FFFu + ((u >> 16) & 1u);          // round-to-nearest-even
  return (unsigned short)(u >> 16);
}

static __device__ __forceinline__ unsigned f32x2_to_bf16x2(float lo, float hi) {
  unsigned a = __float_as_uint(lo);
  unsigned b = __float_as_uint(hi);
  a = (a + 0x7FFFu + ((a >> 16) & 1u)) >> 16;
  b = (b + 0x7FFFu + ((b >> 16) & 1u)) >> 16;
  return (a & 0xFFFFu) | (b << 16);
}

static __device__ __forceinline__ float rowmax16(float v) {
  v = fmaxf(v, __shfl_xor(v, 1));
  v = fmaxf(v, __shfl_xor(v, 2));
  v = fmaxf(v, __shfl_xor(v, 4));
  v = fmaxf(v, __shfl_xor(v, 8));
  return v;
}
static __device__ __forceinline__ float rowsum16(float v) {
  v += __shfl_xor(v, 1);
  v += __shfl_xor(v, 2);
  v += __shfl_xor(v, 4);
  v += __shfl_xor(v, 8);
  return v;
}

// A-frag K-pair index for 16-bit 16x32 A layout (contiguous order, matches B frags)
static __device__ __forceinline__ int a_kpair(int i, int hl) {
  return hl * 16 + 2 * i;
}

// raw 32-bit LDS byte offset of a pointer into __shared__ (for async-LDS VDST)
static __device__ __forceinline__ unsigned lds_offset(const void* p) {
  return (unsigned)(size_t)(__attribute__((address_space(3))) const void*)p;
}

// ---------------------------------------------------------------------------
// Generic GEMM:  C[M=8192, N=1024] = A[M,K=1024] * W^T + bias
// MODE 0: A = fp32 x, store bf16 into head-split [B,H,S,HD] workspace
// MODE 1: A = bf16 O, store fp32 row-major into d_out
// block = 256 thr (8 waves); wave tile 32x64; block tile 128x128
// grid = (M/128, N/128) = (64, 8)
// ---------------------------------------------------------------------------
template <int MODE>
__global__ __launch_bounds__(256) void gemm_wmma_kernel(
    const void* __restrict__ Av, const float* __restrict__ W,
    const float* __restrict__ bias, void* __restrict__ Outv) {
  const int lane = threadIdx.x & 31;
  const int wave = threadIdx.x >> 5;
  const int hl = lane >> 4, nl = lane & 15;
  const int row0 = blockIdx.x * 128 + (wave & 3) * 32;
  const int col0 = blockIdx.y * 128 + (wave >> 2) * 64;

  v8f acc[2][4];
#pragma unroll
  for (int rt = 0; rt < 2; ++rt)
#pragma unroll
    for (int ct = 0; ct < 4; ++ct) acc[rt][ct] = {};

  for (int k0 = 0; k0 < DMODEL; k0 += 32) {
    FragBF a[2];
#pragma unroll
    for (int rt = 0; rt < 2; ++rt) {
      if constexpr (MODE == 0) {
        const float* row = (const float*)Av + (size_t)(row0 + rt * 16 + nl) * DMODEL;
        __builtin_prefetch(row + k0 + 32, 0, 0);   // global_prefetch_b8, next K-slab
#pragma unroll
        for (int i = 0; i < 8; ++i) {
          const int kk = k0 + a_kpair(i, hl);
          const float2 p = *reinterpret_cast<const float2*>(row + kk);
          a[rt].u[i] = f32x2_to_bf16x2(p.x, p.y);
        }
      } else {
        const unsigned short* row =
            (const unsigned short*)Av + (size_t)(row0 + rt * 16 + nl) * DMODEL;
        __builtin_prefetch(row + k0 + 32, 0, 0);
#pragma unroll
        for (int i = 0; i < 8; ++i) {
          const int kk = k0 + a_kpair(i, hl);
          a[rt].u[i] = *reinterpret_cast<const unsigned*>(row + kk);
        }
      }
    }
    FragBF bf[4];
#pragma unroll
    for (int ct = 0; ct < 4; ++ct) {
      // B = W^T, so column n of B is row n of W (contiguous in K)
      const float* wrow = W + (size_t)(col0 + ct * 16 + nl) * DMODEL;
      __builtin_prefetch(wrow + k0 + 32, 0, 0);
#pragma unroll
      for (int i = 0; i < 8; ++i) {
        const int kk = k0 + hl * 16 + 2 * i;
        const float2 p = *reinterpret_cast<const float2*>(wrow + kk);
        bf[ct].u[i] = f32x2_to_bf16x2(p.x, p.y);
      }
    }
#pragma unroll
    for (int rt = 0; rt < 2; ++rt)
#pragma unroll
      for (int ct = 0; ct < 4; ++ct)
        acc[rt][ct] = __builtin_amdgcn_wmma_f32_16x16x32_bf16(
            false, a[rt].v, false, bf[ct].v, (short)0, acc[rt][ct], false, false);
  }

  const int lr8 = (lane < 16) ? 0 : 8;
#pragma unroll
  for (int rt = 0; rt < 2; ++rt) {
#pragma unroll
    for (int ct = 0; ct < 4; ++ct) {
      const int col = col0 + ct * 16 + nl;
      const float bval = bias[col];
#pragma unroll
      for (int r = 0; r < 8; ++r) {
        const int row = row0 + rt * 16 + lr8 + r;
        const float vv = acc[rt][ct][r] + bval;
        if constexpr (MODE == 0) {
          const int b = row >> 11, s = row & (SEQ - 1);
          const int h = col >> 6, hd = col & (HDIM - 1);
          ((unsigned short*)Outv)[(((size_t)(b * NHEAD + h) * SEQ + s) << 6) + hd] =
              f32_to_bf16(vv);
        } else {
          ((float*)Outv)[(size_t)row * DMODEL + col] = vv;
        }
      }
    }
  }
}

// ---------------------------------------------------------------------------
// Flash attention: one wave per 16-query tile; 64-key tiles; causal;
// fp32 online softmax; V tile staged via global_load_async_to_lds_b128
// (ASYNCcnt) overlapped with the QK^T WMMAs; P re-layout through LDS.
// block = 128 thr (4 waves); grid = B*H*(S/16)/4 = 2048 blocks.
// Per-wave LDS: 8 KB V stage + 2 KB P tile -> 40 KB/block.
// ---------------------------------------------------------------------------
__global__ __launch_bounds__(128) void flash_attn_kernel(
    const unsigned short* __restrict__ Qb, const unsigned short* __restrict__ Kb,
    const unsigned short* __restrict__ Vb, unsigned short* __restrict__ Ob) {
  __shared__ __align__(16) unsigned short shmem[4][64 * 64 + 16 * 64];
  const int lane = threadIdx.x & 31;
  const int wave = threadIdx.x >> 5;
  const int hl = lane >> 4, nl = lane & 15;
  const int g = blockIdx.x * 4 + wave;
  const int qt = g & (SEQ / 16 - 1);     // 0..127
  const int bh = g >> 7;                 // 0..63
  const int q0 = qt * 16;
  const unsigned short* Q = Qb + (size_t)bh * SEQ * HDIM;
  const unsigned short* Kp = Kb + (size_t)bh * SEQ * HDIM;
  const unsigned short* Vp = Vb + (size_t)bh * SEQ * HDIM;
  unsigned short* vstage = shmem[wave];            // 64x64 bf16 V tile
  unsigned short* ptile  = shmem[wave] + 64 * 64;  // 16x64 bf16 P tile
  const unsigned vstage_off = lds_offset(vstage);

  // persistent Q fragments (2 K-steps over HD=64)
  FragBF qf[2];
  {
    const unsigned short* row = Q + (size_t)(q0 + nl) * HDIM;
#pragma unroll
    for (int ks = 0; ks < 2; ++ks)
#pragma unroll
      for (int i = 0; i < 8; ++i)
        qf[ks].u[i] =
            *reinterpret_cast<const unsigned*>(row + ks * 32 + a_kpair(i, hl));
  }

  v8f o[4];
#pragma unroll
  for (int u = 0; u < 4; ++u) o[u] = {};
  float mrun[8], lrun[8];
#pragma unroll
  for (int r = 0; r < 8; ++r) { mrun[r] = -1e30f; lrun[r] = 0.f; }

  const int lr8 = (lane < 16) ? 0 : 8;
  const int nkt = ((q0 + 15) >> 6) + 1;

  for (int kt = 0; kt < nkt; ++kt) {
    const int k0 = kt * 64;

    // prior-iteration LDS reads must land before async writes overwrite vstage
    __builtin_amdgcn_wave_barrier();
    asm volatile("s_wait_dscnt 0" ::: "memory");

    // ---- async-stage the 64x64 bf16 V tile (8 KB, contiguous) ----
    {
      const unsigned short* gsrc = Vp + (size_t)k0 * HDIM;
#pragma unroll
      for (int c = 0; c < 16; ++c) {
        const unsigned loff = vstage_off + (unsigned)((c * 32 + lane) * 16);
        const unsigned short* ga = gsrc + (c * 32 + lane) * 8;  // 16 B per lane
        asm volatile("global_load_async_to_lds_b128 %0, %1, off"
                     :: "v"(loff), "v"(ga) : "memory");
      }
    }

    // ---- S = Q K^T (overlaps with the async V stage) ----
    v8f sacc[4];
#pragma unroll
    for (int ct = 0; ct < 4; ++ct) sacc[ct] = {};
#pragma unroll
    for (int ks = 0; ks < 2; ++ks) {
      FragBF kb[4];
#pragma unroll
      for (int ct = 0; ct < 4; ++ct) {
        const unsigned short* krow = Kp + (size_t)(k0 + ct * 16 + nl) * HDIM;
#pragma unroll
        for (int i = 0; i < 8; ++i)
          kb[ct].u[i] =
              *reinterpret_cast<const unsigned*>(krow + ks * 32 + hl * 16 + 2 * i);
      }
#pragma unroll
      for (int ct = 0; ct < 4; ++ct)
        sacc[ct] = __builtin_amdgcn_wmma_f32_16x16x32_bf16(
            false, qf[ks].v, false, kb[ct].v, (short)0, sacc[ct], false, false);
    }

    // ---- causal mask + online softmax (fp32), P -> LDS as bf16 ----
#pragma unroll
    for (int r = 0; r < 8; ++r) {
      const int q = q0 + lr8 + r;
      float rmax = -1e30f;
#pragma unroll
      for (int ct = 0; ct < 4; ++ct) {
        float s = sacc[ct][r] * 0.125f;            // 1/sqrt(64)
        const int key = k0 + ct * 16 + nl;
        if (key > q) s = -1e30f;
        sacc[ct][r] = s;
        rmax = fmaxf(rmax, s);
      }
      rmax = rowmax16(rmax);
      const float mnew = fmaxf(mrun[r], rmax);
      const float alpha = __expf(mrun[r] - mnew);
      mrun[r] = mnew;
      float psum = 0.f;
#pragma unroll
      for (int ct = 0; ct < 4; ++ct) {
        const float p = __expf(sacc[ct][r] - mnew);
        psum += p;
        ptile[(lr8 + r) * 64 + ct * 16 + nl] = f32_to_bf16(p);
      }
      lrun[r] = lrun[r] * alpha + rowsum16(psum);
#pragma unroll
      for (int u = 0; u < 4; ++u) o[u][r] *= alpha;
    }

    __builtin_amdgcn_wave_barrier();
    asm volatile("s_wait_asynccnt 0\n\t"
                 "s_wait_dscnt 0" ::: "memory");    // V staged + P visible

    // ---- O += P V  (B-frags gathered from LDS-staged V) ----
#pragma unroll
    for (int ks = 0; ks < 2; ++ks) {
      FragBF pf;
      {
        const unsigned short* prow = ptile + nl * 64;
#pragma unroll
        for (int i = 0; i < 8; ++i)
          pf.u[i] =
              *reinterpret_cast<const unsigned*>(prow + ks * 32 + a_kpair(i, hl));
      }
      FragBF vf[4];
#pragma unroll
      for (int u = 0; u < 4; ++u) {
#pragma unroll
        for (int i = 0; i < 8; ++i) {
          const int key = ks * 32 + hl * 16 + 2 * i;
          const unsigned s0 = vstage[key * 64 + u * 16 + nl];
          const unsigned s1 = vstage[(key + 1) * 64 + u * 16 + nl];
          vf[u].u[i] = s0 | (s1 << 16);
        }
      }
#pragma unroll
      for (int u = 0; u < 4; ++u)
        o[u] = __builtin_amdgcn_wmma_f32_16x16x32_bf16(
            false, pf.v, false, vf[u].v, (short)0, o[u], false, false);
    }
    __builtin_amdgcn_wave_barrier();
  }

  // ---- normalize and store O as bf16 in [B,S,D] (merged heads) ----
  const int b = bh >> 4, h = bh & (NHEAD - 1);
#pragma unroll
  for (int r = 0; r < 8; ++r) {
    const int q = q0 + lr8 + r;
    const float inv = 1.0f / lrun[r];
#pragma unroll
    for (int u = 0; u < 4; ++u) {
      const int col = h * HDIM + u * 16 + nl;
      Ob[(size_t)(b * SEQ + q) * DMODEL + col] = f32_to_bf16(o[u][r] * inv);
    }
  }
}

// ---------------------------------------------------------------------------
extern "C" void kernel_launch(void* const* d_in, const int* in_sizes, int n_in,
                              void* d_out, int out_size, void* d_ws, size_t ws_size,
                              hipStream_t stream) {
  const float* x  = (const float*)d_in[0];
  const float* Wq = (const float*)d_in[1];
  const float* bq = (const float*)d_in[2];
  const float* Wk = (const float*)d_in[3];
  const float* bk = (const float*)d_in[4];
  const float* Wv = (const float*)d_in[5];
  const float* bv = (const float*)d_in[6];
  const float* Wo = (const float*)d_in[7];
  const float* bo = (const float*)d_in[8];

  const size_t NE = (size_t)BATCH * NHEAD * SEQ * HDIM;  // 8388608 elems (bf16 -> 16 MB)
  unsigned short* Qb = (unsigned short*)d_ws;
  unsigned short* Kb = Qb + NE;
  unsigned short* Vb = Kb + NE;
  unsigned short* Ob = Vb + NE;   // total 64 MB of workspace

  dim3 gproj(BATCH * SEQ / 128, DMODEL / 128);  // (64, 8)
  gemm_wmma_kernel<0><<<gproj, dim3(256), 0, stream>>>(x, Wq, bq, Qb);
  gemm_wmma_kernel<0><<<gproj, dim3(256), 0, stream>>>(x, Wk, bk, Kb);
  gemm_wmma_kernel<0><<<gproj, dim3(256), 0, stream>>>(x, Wv, bv, Vb);

  const int nwaves = BATCH * NHEAD * (SEQ / 16);          // 8192
  flash_attn_kernel<<<nwaves / 4, dim3(128), 0, stream>>>(Qb, Kb, Vb, Ob);

  gemm_wmma_kernel<1><<<gproj, dim3(256), 0, stream>>>(Ob, Wo, bo, (float*)d_out);
}
